// AttentionBlock_6597069767006
// MI455X (gfx1250) — compile-verified, matrix operand order checked by reading, not run on hardware
//
#include <hip/hip_runtime.h>
#include <stdint.h>

#define BATCH 8
#define CH 256
#define NPIX 4096
#define NGROUPS 8
#define CPG (CH / NGROUPS)
#define EPS 1e-5f

// LDS strides (elements), multiples of 8 so 8-element runs are 16B aligned
#define CSTR 264   // [n][c] tiles, c-stride padded
#define KSTR 264   // Ks [key][c]
#define VSTR 40    // Vs [c][key], 32 keys + pad

typedef __attribute__((ext_vector_type(16))) __bf16 v16bf;
typedef __attribute__((ext_vector_type(8)))  float  v8f;
typedef __attribute__((ext_vector_type(4)))  int    v4i_;

#if defined(__AMDGCN__) && __has_builtin(__builtin_amdgcn_global_load_async_to_lds_b128) && __has_builtin(__builtin_amdgcn_s_wait_asynccnt)
#define USE_ASYNC 1
#else
#define USE_ASYNC 0
#endif

// async builtin wants: (global int4*, lds int4*, i32 offset, i32 cpol)
#define ASG(p) ((__attribute__((address_space(1))) v4i_*)(p))
#define ASL(p) ((__attribute__((address_space(3))) v4i_*)(p))

union FragPack {
    v16bf v;
    unsigned short u[16];
    uint4 q[2];
};

__device__ __forceinline__ unsigned short f2bf(float x) {
    __bf16 h = (__bf16)x;              // native cvt, round-to-nearest-even
    return __builtin_bit_cast(unsigned short, h);
}

__device__ __forceinline__ v8f vzero8() {
    v8f z;
#pragma unroll
    for (int i = 0; i < 8; ++i) z[i] = 0.0f;
    return z;
}

// Load a 16-bit A/B fragment: per-lane data = two contiguous 8-element runs at
// base and base+16 (ISA layout: K = half*8 + 0..7 and 16 + half*8 + 0..7)
__device__ __forceinline__ v16bf ldFrag(const unsigned short* base) {
    FragPack f;
    f.q[0] = *reinterpret_cast<const uint4*>(base);
    f.q[1] = *reinterpret_cast<const uint4*>(base + 16);
    return f.v;
}

__device__ __forceinline__ v8f wmma_bf16(v16bf a, v16bf b, v8f c) {
    return __builtin_amdgcn_wmma_f32_16x16x32_bf16(false, a, false, b,
                                                   (short)0, c, false, false);
}

// ---------------------------------------------------------------------------
// Kernel 0: convert weights to bf16 once
// ---------------------------------------------------------------------------
__global__ void cvtw_kernel(const float* __restrict__ wq,
                            const float* __restrict__ wp,
                            unsigned short* __restrict__ wqb,
                            unsigned short* __restrict__ wpb) {
    int i = blockIdx.x * 256 + threadIdx.x;
    if (i < 3 * CH * CH) wqb[i] = f2bf(wq[i]);
    if (i < CH * CH)     wpb[i] = f2bf(wp[i]);
}

// ---------------------------------------------------------------------------
// Kernel 1: GroupNorm -> bf16 activations [b][c][n]
// ---------------------------------------------------------------------------
__global__ void gn_kernel(const float* __restrict__ x,
                          const float* __restrict__ scale,
                          const float* __restrict__ bias,
                          unsigned short* __restrict__ xn) {
    int b = blockIdx.y, g = blockIdx.x;
    int tid = threadIdx.x;
    const int NE = CPG * NPIX;   // 131072
    const int NV = NE / 4;
    const float* xp = x + ((size_t)b * CH + (size_t)g * CPG) * NPIX;

    float s = 0.f, ss = 0.f;
    for (int i = tid; i < NV; i += 256) {
        float4 v = ((const float4*)xp)[i];
        s += v.x + v.y + v.z + v.w;
        ss += v.x * v.x + v.y * v.y + v.z * v.z + v.w * v.w;
    }
    __shared__ float rs[256], rss[256];
    rs[tid] = s; rss[tid] = ss;
    __syncthreads();
    for (int o = 128; o > 0; o >>= 1) {
        if (tid < o) { rs[tid] += rs[tid + o]; rss[tid] += rss[tid + o]; }
        __syncthreads();
    }
    float mean = rs[0] * (1.0f / NE);
    float var  = rss[0] * (1.0f / NE) - mean * mean;
    float rstd = rsqrtf(var + EPS);

    unsigned short* xo = xn + ((size_t)b * CH + (size_t)g * CPG) * NPIX;
    for (int i = tid; i < NV; i += 256) {
        int c = g * CPG + (i >> 10);
        float sc = scale[c] * rstd;
        float bi = bias[c] - mean * sc;
        float4 v = ((const float4*)xp)[i];
        ushort4 o4;
        o4.x = f2bf(v.x * sc + bi);
        o4.y = f2bf(v.y * sc + bi);
        o4.z = f2bf(v.z * sc + bi);
        o4.w = f2bf(v.w * sc + bi);
        ((ushort4*)xo)[i] = o4;
    }
}

// ---------------------------------------------------------------------------
// Kernel 2: QKV GEMM; writes Q, K, V all as [b][n][c] bf16
// ---------------------------------------------------------------------------
__global__ void qkv_kernel(const unsigned short* __restrict__ xn,
                           const unsigned short* __restrict__ wqb,
                           const float* __restrict__ bvec,
                           unsigned short* __restrict__ Qb,
                           unsigned short* __restrict__ Kb,
                           unsigned short* __restrict__ Vb) {
    __shared__ unsigned short Xs[64 * CSTR];  // [n][c]
    int b = blockIdx.z;
    int otile = blockIdx.y;
    int ntile = blockIdx.x * 64;
    int tid = threadIdx.x;
    int wave = tid >> 5, lane = tid & 31;
    int half = lane >> 4, col = lane & 15;

    const unsigned short* xp = xn + (size_t)b * CH * NPIX;
    for (int idx = tid; idx < 64 * CH; idx += 128) {
        int n = idx & 63, c = idx >> 6;
        Xs[n * CSTR + c] = xp[(size_t)c * NPIX + ntile + n];
    }
    __syncthreads();

    int orow = otile * 64 + wave * 16 + col;
    v16bf af[8];
#pragma unroll
    for (int k = 0; k < 8; ++k)
        af[k] = ldFrag(wqb + (size_t)orow * CH + k * 32 + half * 8);

    v8f acc[4];
#pragma unroll
    for (int s = 0; s < 4; ++s) acc[s] = vzero8();

#pragma unroll
    for (int s = 0; s < 4; ++s) {
        int n = s * 16 + col;
#pragma unroll
        for (int k = 0; k < 8; ++k) {
            v16bf bf = ldFrag(&Xs[n * CSTR + k * 32 + half * 8]);
            acc[s] = wmma_bf16(af[k], bf, acc[s]);
        }
    }

#pragma unroll
    for (int s = 0; s < 4; ++s) {
        int n = ntile + s * 16 + col;
#pragma unroll
        for (int r = 0; r < 8; ++r) {
            int o = otile * 64 + wave * 16 + r + 8 * half;
            unsigned short bfv = f2bf(acc[s][r] + bvec[o]);
            if (o < CH) {
                Qb[((size_t)b * NPIX + n) * CH + o] = bfv;
            } else if (o < 2 * CH) {
                Kb[((size_t)b * NPIX + n) * CH + (o - CH)] = bfv;
            } else {
                Vb[((size_t)b * NPIX + n) * CH + (o - 2 * CH)] = bfv;
            }
        }
    }
}

// ---------------------------------------------------------------------------
// Kernel 3: flash-style attention, 4 waves x 16 query rows
// ---------------------------------------------------------------------------
__global__ void attn_kernel(const unsigned short* __restrict__ Qb,
                            const unsigned short* __restrict__ Kb,
                            const unsigned short* __restrict__ Vb,
                            unsigned short* __restrict__ Ob) {
    __shared__ unsigned short Ks[32 * KSTR];   // [key][c]
    __shared__ unsigned short Vs[CH * VSTR];   // [c][key]
    __shared__ unsigned short Ps[4][16 * 32];  // per-wave P tile [row][key]

    int b = blockIdx.y;
    int qtile = blockIdx.x * 64;
    int tid = threadIdx.x, wave = tid >> 5, lane = tid & 31;
    int half = lane >> 4, col = lane & 15;
    const float sm_scale = 0.0625f;  // 1/sqrt(256)

    const unsigned short* qp = Qb + (size_t)b * NPIX * CH;
    const unsigned short* kp = Kb + (size_t)b * NPIX * CH;
    const unsigned short* vp = Vb + (size_t)b * NPIX * CH;

    int qrow = qtile + wave * 16 + col;
    v16bf qf[8];
#pragma unroll
    for (int k = 0; k < 8; ++k)
        qf[k] = ldFrag(qp + (size_t)qrow * CH + k * 32 + half * 8);

    v8f oacc[16];
#pragma unroll
    for (int t = 0; t < 16; ++t) oacc[t] = vzero8();
    float rm[8], rl[8];
#pragma unroll
    for (int r = 0; r < 8; ++r) { rm[r] = -1e30f; rl[r] = 0.0f; }

    for (int jb = 0; jb < NPIX; jb += 32) {
        __syncthreads();

        int prow = jb + 32 + (tid >> 2);
        if (prow < NPIX) {
            __builtin_prefetch(kp + (size_t)prow * CH + (tid & 3) * 64, 0, 1);
            __builtin_prefetch(vp + (size_t)prow * CH + (tid & 3) * 64, 0, 1);
        }

#if USE_ASYNC
        for (int idx = tid; idx < 1024; idx += 128) {
            int key = idx >> 5, ch8 = (idx & 31) * 8;
            __builtin_amdgcn_global_load_async_to_lds_b128(
                ASG(kp + (size_t)(jb + key) * CH + ch8),
                ASL(&Ks[key * KSTR + ch8]), 0, 0);
        }
#else
        for (int idx = tid; idx < 1024; idx += 128) {
            int key = idx >> 5, ch8 = (idx & 31) * 8;
            *(uint4*)&Ks[key * KSTR + ch8] =
                *(const uint4*)(kp + (size_t)(jb + key) * CH + ch8);
        }
#endif
        for (int idx = tid; idx < 1024; idx += 128) {
            int key = idx >> 5, ch8 = (idx & 31) * 8;
            uint4 d = *(const uint4*)(vp + (size_t)(jb + key) * CH + ch8);
            const unsigned short* e = (const unsigned short*)&d;
#pragma unroll
            for (int j = 0; j < 8; ++j)
                Vs[(ch8 + j) * VSTR + key] = e[j];
        }
#if USE_ASYNC
        __builtin_amdgcn_s_wait_asynccnt(0);
#endif
        __syncthreads();

        v8f s0 = vzero8(), s1 = vzero8();
#pragma unroll
        for (int k = 0; k < 8; ++k) {
            v16bf b0 = ldFrag(&Ks[col * KSTR + k * 32 + half * 8]);
            s0 = wmma_bf16(qf[k], b0, s0);
        }
#pragma unroll
        for (int k = 0; k < 8; ++k) {
            v16bf b1 = ldFrag(&Ks[(16 + col) * KSTR + k * 32 + half * 8]);
            s1 = wmma_bf16(qf[k], b1, s1);
        }

#pragma unroll
        for (int r = 0; r < 8; ++r) {
            float a0 = s0[r] * sm_scale;
            float a1 = s1[r] * sm_scale;
            float mx = fmaxf(a0, a1);
#pragma unroll
            for (int m = 8; m >= 1; m >>= 1)
                mx = fmaxf(mx, __shfl_xor(mx, m, 32));
            float mnew = fmaxf(rm[r], mx);
            float alpha = __expf(rm[r] - mnew);
            float p0 = __expf(a0 - mnew);
            float p1 = __expf(a1 - mnew);
            float psum = p0 + p1;
#pragma unroll
            for (int m = 8; m >= 1; m >>= 1)
                psum += __shfl_xor(psum, m, 32);
            rl[r] = rl[r] * alpha + psum;
            rm[r] = mnew;
#pragma unroll
            for (int t = 0; t < 16; ++t) oacc[t][r] *= alpha;
            int row = r + 8 * half;
            Ps[wave][row * 32 + col]      = f2bf(p0);
            Ps[wave][row * 32 + 16 + col] = f2bf(p1);
        }
        __syncthreads();

        v16bf pa = ldFrag(&Ps[wave][col * 32 + half * 8]);

#pragma unroll
        for (int t = 0; t < 16; ++t) {
            v16bf bv = ldFrag(&Vs[(t * 16 + col) * VSTR + half * 8]);
            oacc[t] = wmma_bf16(pa, bv, oacc[t]);
        }
    }

#pragma unroll
    for (int r = 0; r < 8; ++r) {
        float inv = 1.0f / rl[r];
        int row = qtile + wave * 16 + r + 8 * half;
#pragma unroll
        for (int t = 0; t < 16; ++t) {
            Ob[((size_t)b * NPIX + row) * CH + t * 16 + col] =
                f2bf(oacc[t][r] * inv);
        }
    }
}

// ---------------------------------------------------------------------------
// Kernel 4: proj GEMM + bias + residual -> f32 output
// ---------------------------------------------------------------------------
__global__ void proj_kernel(const unsigned short* __restrict__ Ab,
                            const unsigned short* __restrict__ wpb,
                            const float* __restrict__ bvec,
                            const float* __restrict__ x,
                            float* __restrict__ out) {
    __shared__ unsigned short Bs[64 * CSTR];  // [n][c]
    int b = blockIdx.z;
    int otile = blockIdx.y * 64;
    int ntile = blockIdx.x * 64;
    int tid = threadIdx.x;
    int wave = tid >> 5, lane = tid & 31;
    int half = lane >> 4, col = lane & 15;

    const unsigned short* ap = Ab + (size_t)b * NPIX * CH;
#if USE_ASYNC
    for (int idx = tid; idx < 2048; idx += 128) {
        int n = idx >> 5, ch8 = (idx & 31) * 8;
        __builtin_amdgcn_global_load_async_to_lds_b128(
            ASG(ap + (size_t)(ntile + n) * CH + ch8),
            ASL(&Bs[n * CSTR + ch8]), 0, 0);
    }
    __builtin_amdgcn_s_wait_asynccnt(0);
#else
    for (int idx = tid; idx < 2048; idx += 128) {
        int n = idx >> 5, ch8 = (idx & 31) * 8;
        *(uint4*)&Bs[n * CSTR + ch8] =
            *(const uint4*)(ap + (size_t)(ntile + n) * CH + ch8);
    }
#endif
    __syncthreads();

    int orow = otile + wave * 16 + col;
    v16bf af[8];
#pragma unroll
    for (int k = 0; k < 8; ++k)
        af[k] = ldFrag(wpb + (size_t)orow * CH + k * 32 + half * 8);

    v8f acc[4];
#pragma unroll
    for (int s = 0; s < 4; ++s) acc[s] = vzero8();

#pragma unroll
    for (int s = 0; s < 4; ++s) {
        int n = s * 16 + col;
#pragma unroll
        for (int k = 0; k < 8; ++k) {
            v16bf bf = ldFrag(&Bs[n * CSTR + k * 32 + half * 8]);
            acc[s] = wmma_bf16(af[k], bf, acc[s]);
        }
    }

#pragma unroll
    for (int s = 0; s < 4; ++s) {
        int n = ntile + s * 16 + col;
#pragma unroll
        for (int r = 0; r < 8; ++r) {
            int o = otile + wave * 16 + r + 8 * half;
            size_t gi = ((size_t)b * CH + o) * NPIX + n;
            out[gi] = x[gi] + acc[s][r] + bvec[o];
        }
    }
}

// ---------------------------------------------------------------------------
extern "C" void kernel_launch(void* const* d_in, const int* in_sizes, int n_in,
                              void* d_out, int out_size, void* d_ws, size_t ws_size,
                              hipStream_t stream) {
    (void)in_sizes; (void)n_in; (void)out_size; (void)ws_size;
    const float* x     = (const float*)d_in[0];
    const float* gns   = (const float*)d_in[1];
    const float* gnb   = (const float*)d_in[2];
    const float* qkvw  = (const float*)d_in[3];
    const float* qkvb  = (const float*)d_in[4];
    const float* projw = (const float*)d_in[5];
    const float* projb = (const float*)d_in[6];
    float* out = (float*)d_out;

    char* ws = (char*)d_ws;
    const size_t seg = (size_t)BATCH * CH * NPIX * sizeof(unsigned short);
    unsigned short* xn  = (unsigned short*)(ws);
    unsigned short* Qb  = (unsigned short*)(ws + seg);
    unsigned short* Kb  = (unsigned short*)(ws + 2 * seg);
    unsigned short* Vb  = (unsigned short*)(ws + 3 * seg);
    unsigned short* Ob  = (unsigned short*)(ws + 4 * seg);
    unsigned short* wqb = (unsigned short*)(ws + 5 * seg);
    unsigned short* wpb = wqb + (size_t)3 * CH * CH;

    cvtw_kernel<<<(3 * CH * CH + 255) / 256, 256, 0, stream>>>(qkvw, projw, wqb, wpb);
    gn_kernel<<<dim3(NGROUPS, BATCH), 256, 0, stream>>>(x, gns, gnb, xn);
    qkv_kernel<<<dim3(NPIX / 64, (3 * CH) / 64, BATCH), 128, 0, stream>>>(
        xn, wqb, qkvb, Qb, Kb, Vb);
    attn_kernel<<<dim3(NPIX / 64, BATCH), 128, 0, stream>>>(Qb, Kb, Vb, Ob);
    proj_kernel<<<dim3(NPIX / 64, CH / 64, BATCH), 128, 0, stream>>>(
        Ob, wpb, projb, x, out);
}
